// BayesianDenseLayer_57595511439819
// MI455X (gfx1250) — matrix operand order repaired and checked by reading, not compiled, not run on hardware
//
#include <hip/hip_runtime.h>
#include <hip/hip_bf16.h>

typedef __attribute__((ext_vector_type(16))) _Float16 v16h;
typedef __attribute__((ext_vector_type(8)))  float    v8f;

union Frag16 { v16h v; uint4 q[2]; };

__device__ __forceinline__ float softplusf(float v) {
    return fmaxf(v, 0.0f) + log1pf(__expf(-fabsf(v)));
}

// CDNA5 async global->LDS copy (ASYNCcnt-tracked), GVS addressing:
// mem = SGPR base + VGPR 32-bit byte offset; dst = LDS byte address in VGPR.
__device__ __forceinline__ void async_b128(unsigned lds_addr, const void* base,
                                           unsigned goff) {
    asm volatile("global_load_async_to_lds_b128 %0, %1, %2"
                 :: "v"(lds_addr), "v"(goff), "s"(base)
                 : "memory");
}
__device__ __forceinline__ void wait_async0() {
    asm volatile("s_wait_asynccnt 0x0" ::: "memory");
}

// ---------------------------------------------------------------------------
// Pass 1a: x (f32, [M,K]) -> xh (f16, [M,K])
// ---------------------------------------------------------------------------
__global__ __launch_bounds__(256) void bdl_conv_x(const float* __restrict__ x,
                                                  _Float16* __restrict__ xh) {
    size_t i = (size_t)blockIdx.x * blockDim.x + threadIdx.x;   // float4 index
    float4 v = ((const float4*)x)[i];
    union { _Float16 h[4]; uint2 q; } o;
    o.h[0] = (_Float16)v.x; o.h[1] = (_Float16)v.y;
    o.h[2] = (_Float16)v.z; o.h[3] = (_Float16)v.w;
    ((uint2*)xh)[i] = o.q;
}

// ---------------------------------------------------------------------------
// Pass 1b: W = w_loc + softplus(w_std)*eps_w, f32 [K,N] -> f16 TRANSPOSED [N,K]
// ---------------------------------------------------------------------------
__global__ __launch_bounds__(256) void bdl_make_wT(const float* __restrict__ wl,
                                                   const float* __restrict__ wstd,
                                                   const float* __restrict__ ew,
                                                   _Float16* __restrict__ wT) {
    const int N = 4096, K = 4096;
    __shared__ _Float16 tile[64][72];

    int tid = threadIdx.x;
    int r   = tid >> 2;
    int cs  = (tid & 3) << 4;
    int k0  = blockIdx.y << 6;
    int n0  = blockIdx.x << 6;

    size_t base = (size_t)(k0 + r) * N + (n0 + cs);
#pragma unroll
    for (int i = 0; i < 4; ++i) {
        float4 a = *(const float4*)(wl   + base + 4 * i);
        float4 s = *(const float4*)(wstd + base + 4 * i);
        float4 e = *(const float4*)(ew   + base + 4 * i);
        tile[r][cs + 4 * i + 0] = (_Float16)(a.x + softplusf(s.x) * e.x);
        tile[r][cs + 4 * i + 1] = (_Float16)(a.y + softplusf(s.y) * e.y);
        tile[r][cs + 4 * i + 2] = (_Float16)(a.z + softplusf(s.z) * e.z);
        tile[r][cs + 4 * i + 3] = (_Float16)(a.w + softplusf(s.w) * e.w);
    }
    __syncthreads();

    union { _Float16 h[16]; uint4 q[2]; } o;
#pragma unroll
    for (int i = 0; i < 16; ++i) o.h[i] = tile[cs + i][r];
    size_t ob = (size_t)(n0 + r) * K + (k0 + cs);
    *(uint4*)(wT + ob)     = o.q[0];
    *(uint4*)(wT + ob + 8) = o.q[1];
}

// ---------------------------------------------------------------------------
// Pass 1c: bias
// ---------------------------------------------------------------------------
__global__ __launch_bounds__(256) void bdl_make_bias(const float* __restrict__ bl,
                                                     const float* __restrict__ bs,
                                                     const float* __restrict__ eb,
                                                     float* __restrict__ bias) {
    int i = blockIdx.x * blockDim.x + threadIdx.x;
    bias[i] = bl[i] + softplusf(bs[i]) * eb[i];
}

// ---------------------------------------------------------------------------
// Pass 2: C = A * Bt^T + bias via v_wmma_f32_16x16x32_f16.
// Block tile 128x128, BK=64 (2 wmma k-steps / LDS tile), 8 waves x (64x32).
// Double-buffered LDS fed by GLOBAL_LOAD_ASYNC_TO_LDS_B128 (ASYNCcnt),
// one barrier per k-tile.
// ---------------------------------------------------------------------------
__global__ __launch_bounds__(256) void bdl_wmma_gemm(const _Float16* __restrict__ A,   // [M,K]
                                                     const _Float16* __restrict__ Bt,  // [N,K]
                                                     const float* __restrict__ bias,   // [N]
                                                     float* __restrict__ C) {          // [M,N]
    const int K  = 4096;
    const int Nn = 4096;
    const int KT = K / 64;                      // 64 k-tiles

    __shared__ __align__(16) _Float16 As[2][128][72];   // rows 144B (16B-aligned)
    __shared__ __align__(16) _Float16 Bs[2][128][72];
    const unsigned BUFSZ = 128 * 72 * 2;        // 18432 B per buffer

    int tid  = threadIdx.x;
    int lane = tid & 31;
    int wid  = tid >> 5;
    int wm   = wid >> 2;                        // 0..1
    int wn   = wid & 3;                         // 0..3
    int m0   = blockIdx.y * 128;
    int n0   = blockIdx.x * 128;

    // staging: thread owns 64B (4 x b128) of one row-half; 256 thr = 128 rows x 128B
    int sr = tid >> 1;                          // 0..127
    int sc = (tid & 1) * 64;                    // byte offset in row: 0 / 64
    unsigned aoff  = (unsigned)(m0 + sr) * (unsigned)(K * 2) + (unsigned)sc;
    unsigned boff  = (unsigned)(n0 + sr) * (unsigned)(K * 2) + (unsigned)sc;
    unsigned ldsA0 = (unsigned)(size_t)&As[0][0][0] + (unsigned)(sr * 144 + sc);
    unsigned ldsB0 = (unsigned)(size_t)&Bs[0][0][0] + (unsigned)(sr * 144 + sc);

    v8f acc[4][2];
#pragma unroll
    for (int i = 0; i < 4; ++i)
#pragma unroll
        for (int j = 0; j < 2; ++j) acc[i][j] = (v8f){0.f,0.f,0.f,0.f,0.f,0.f,0.f,0.f};

    // prologue: async-stage k-tile 0 into buffer 0
#pragma unroll
    for (int i = 0; i < 4; ++i) {
        async_b128(ldsA0 + 16 * i, A,  aoff + 16 * i);
        async_b128(ldsB0 + 16 * i, Bt, boff + 16 * i);
    }
    wait_async0();
    __syncthreads();

    int half = lane >> 4;
    int l16  = lane & 15;

    int buf = 0;
    for (int kt = 0; kt < KT; ++kt) {
        if (kt + 1 < KT) {                      // async-stage next tile -> back buffer
            unsigned kadd = (unsigned)(kt + 1) * 128u;   // 64 halves per tile
            unsigned la = ldsA0 + (unsigned)(buf ^ 1) * BUFSZ;
            unsigned lb = ldsB0 + (unsigned)(buf ^ 1) * BUFSZ;
#pragma unroll
            for (int i = 0; i < 4; ++i) {
                async_b128(la + 16 * i, A,  aoff + kadd + 16 * i);
                async_b128(lb + 16 * i, Bt, boff + kadd + 16 * i);
            }
        }

        const _Float16* Abase = &As[buf][0][0];
        const _Float16* Bbase = &Bs[buf][0][0];
#pragma unroll
        for (int kk = 0; kk < 2; ++kk) {
            // B fragments: lane = column n, contiguous K halves [kk*32+half*16, +16)
            Frag16 bf[2];
#pragma unroll
            for (int tn = 0; tn < 2; ++tn) {
                const _Float16* p = Bbase + (wn * 32 + tn * 16 + l16) * 72
                                          + kk * 32 + half * 16;
                bf[tn].q[0] = *(const uint4*)p;
                bf[tn].q[1] = *(const uint4*)(p + 8);
            }
            // A fragments: contiguous K halves [kk*32+half*16, +16)
#pragma unroll
            for (int tm = 0; tm < 4; ++tm) {
                const _Float16* p = Abase + (wm * 64 + tm * 16 + l16) * 72
                                          + kk * 32 + half * 16;
                Frag16 af;
                af.q[0] = *(const uint4*)p;
                af.q[1] = *(const uint4*)(p + 8);
                acc[tm][0] = __builtin_amdgcn_wmma_f32_16x16x32_f16(
                    false, af.v, false, bf[0].v, (short)0, acc[tm][0], false, false);
                acc[tm][1] = __builtin_amdgcn_wmma_f32_16x16x32_f16(
                    false, af.v, false, bf[1].v, (short)0, acc[tm][1], false, false);
            }
        }

        wait_async0();          // next tile has landed in LDS
        __syncthreads();        // everyone done reading front buffer
        buf ^= 1;
    }

    // epilogue: C/D layout -> lane 0-15: M=r, N=lane; lane 16-31: M=8+r, N=lane-16
    int mh = (lane >> 4) * 8;
#pragma unroll
    for (int tn = 0; tn < 2; ++tn) {
        int n = n0 + wn * 32 + tn * 16 + l16;
        float bv = bias[n];
#pragma unroll
        for (int tm = 0; tm < 4; ++tm) {
            int mbase = m0 + wm * 64 + tm * 16 + mh;
#pragma unroll
            for (int r2 = 0; r2 < 8; ++r2) {
                C[(size_t)(mbase + r2) * Nn + n] = acc[tm][tn][r2] + bv;
            }
        }
    }
}

// ---------------------------------------------------------------------------
extern "C" void kernel_launch(void* const* d_in, const int* in_sizes, int n_in,
                              void* d_out, int out_size, void* d_ws, size_t ws_size,
                              hipStream_t stream) {
    const float* x     = (const float*)d_in[0];
    const float* w_loc = (const float*)d_in[1];
    const float* w_std = (const float*)d_in[2];
    const float* b_loc = (const float*)d_in[3];
    const float* b_std = (const float*)d_in[4];
    const float* eps_w = (const float*)d_in[5];
    const float* eps_b = (const float*)d_in[6];
    float* out = (float*)d_out;

    const int M = 8192, N = 4096, K = 4096;

    _Float16* xh   = (_Float16*)d_ws;
    _Float16* wT   = xh + (size_t)M * K;
    float*    bias = (float*)(wT + (size_t)N * K);

    bdl_conv_x<<<(M * (size_t)K) / (256 * 4), 256, 0, stream>>>(x, xh);
    bdl_make_wT<<<dim3(N / 64, K / 64), 256, 0, stream>>>(w_loc, w_std, eps_w, wT);
    bdl_make_bias<<<N / 256, 256, 0, stream>>>(b_loc, b_std, eps_b, bias);
    bdl_wmma_gemm<<<dim3(N / 128, M / 128), 256, 0, stream>>>(xh, wT, bias, out);
}
